// GAT_42975442764234
// MI455X (gfx1250) — matrix operand order repaired and checked by reading, not compiled, not run on hardware
//
#include <hip/hip_runtime.h>
#include <hip/hip_bf16.h>
#include <stdint.h>

#define NEG_SLOPE 0.2f

typedef __attribute__((ext_vector_type(16))) __bf16 bf16x16;
typedef __attribute__((ext_vector_type(8)))  __bf16 bf16x8;
typedef __attribute__((ext_vector_type(8)))  float  f32x8;

// ---- float <-> order-preserving int (for atomicMax on float) ----
__device__ __forceinline__ int f2ord(float f) {
    int i = __float_as_int(f);
    return i >= 0 ? i : (i ^ 0x7fffffff);
}
__device__ __forceinline__ float ord2f(int i) {
    return __int_as_float(i >= 0 ? i : (i ^ 0x7fffffff));
}

// ================= conversion kernels =================
__global__ void cvt_x_kernel(const float* __restrict__ x, __bf16* __restrict__ xb,
                             int n, int npad) {
    size_t t = (size_t)blockIdx.x * blockDim.x + threadIdx.x;
    size_t total = (size_t)npad * 256;
    if (t >= total) return;
    int row = (int)(t >> 8);
    float v = (row < n) ? x[t] : 0.0f;
    xb[t] = (__bf16)v;
}

// W [256, fout] row-major (f32)  ->  Wt [fout, 256] (bf16), i.e. transposed
__global__ void cvt_wt_kernel(const float* __restrict__ W, __bf16* __restrict__ Wt, int fout) {
    int t = blockIdx.x * blockDim.x + threadIdx.x;
    if (t >= fout * 256) return;
    int j = t >> 8;      // output feature
    int k = t & 255;     // input feature
    Wt[t] = (__bf16)W[(size_t)k * fout + j];
}

// ================= WMMA GEMM: H[n,fout] = Xb[n,256] @ Wt^T =================
// Block = 256 threads = 8 waves. Wave w: M-tile (w&3), N-tile pair (w>>2).
// Block tile: 64 rows x 64 cols. K = 256, stepped by 32 (bf16 WMMA).
__global__ __launch_bounds__(256)
void gemm_wmma_kernel(const __bf16* __restrict__ Xb,   // [npad, 256]
                      const __bf16* __restrict__ Wt,   // [fout, 256] (transposed weight)
                      float* __restrict__ Hout,        // [n, fout]
                      int n, int fout) {
    const int lane = threadIdx.x & 31;
    const int wave = threadIdx.x >> 5;
    const int mt   = wave & 3;
    const int np   = wave >> 2;
    const int row0 = blockIdx.x * 64 + mt * 16;
    const int col0 = blockIdx.y * 64 + np * 32;
    const int half = lane >> 4;     // lane half: selects K sub-range 16*half..16*half+15 (contiguous order)
    const int mr   = lane & 15;     // A: row within M-tile;  B/D: column within N-tile

    f32x8 acc0 = {};
    f32x8 acc1 = {};

    // A fragment: lane reads two 16B chunks at k+16*half and k+16*half+8 (K contiguous, matches B)
    const __bf16* xr = Xb + (size_t)(row0 + mr) * 256 + 16 * half;
    // B fragment: lanes 0-15 K=0..15, lanes 16-31 K=16..31, contiguous in Wt row
    const __bf16* w0 = Wt + (size_t)(col0 + mr) * 256 + 16 * half;
    const __bf16* w1 = w0 + (size_t)16 * 256;

#pragma unroll
    for (int kk = 0; kk < 8; ++kk) {
        const int k0 = kk * 32;
        bf16x8 alo = *(const bf16x8*)(xr + k0);
        bf16x8 ahi = *(const bf16x8*)(xr + k0 + 8);
        union { bf16x16 v; bf16x8 h[2]; } ua;
        ua.h[0] = alo; ua.h[1] = ahi;
        bf16x16 b0 = *(const bf16x16*)(w0 + k0);
        bf16x16 b1 = *(const bf16x16*)(w1 + k0);
        acc0 = __builtin_amdgcn_wmma_f32_16x16x32_bf16(false, ua.v, false, b0,
                                                       (short)0, acc0, false, false);
        acc1 = __builtin_amdgcn_wmma_f32_16x16x32_bf16(false, ua.v, false, b1,
                                                       (short)0, acc1, false, false);
    }

    // D layout: VGPR r -> row M = r + 8*(lane>>4); column N = lane&15
#pragma unroll
    for (int r = 0; r < 8; ++r) {
        int row = row0 + r + 8 * half;
        if (row < n) {
            Hout[(size_t)row * fout + col0 + mr]      = acc0[r];
            Hout[(size_t)row * fout + col0 + 16 + mr] = acc1[r];
        }
    }
}

// ================= attention coefficients =================
__global__ void attn_coef_kernel(const float* __restrict__ h,
                                 const float* __restrict__ a_src,
                                 const float* __restrict__ a_dst,
                                 float* __restrict__ als, float* __restrict__ ald,
                                 int n, int Hh, int C) {
    int t = blockIdx.x * blockDim.x + threadIdx.x;
    if (t >= n * Hh) return;
    int node = t / Hh, hd = t - node * Hh;
    const float* hr = h + (size_t)node * Hh * C + (size_t)hd * C;
    const float* s  = a_src + hd * C;
    const float* d  = a_dst + hd * C;
    float ss = 0.f, sd = 0.f;
    for (int c = 0; c < C; ++c) {
        float v = hr[c];
        ss = fmaf(v, s[c], ss);
        sd = fmaf(v, d[c], sd);
    }
    als[t] = ss;
    ald[t] = sd;
}

// ================= fills =================
__global__ void fill_f32_kernel(float* __restrict__ p, float v, size_t count) {
    size_t t = (size_t)blockIdx.x * blockDim.x + threadIdx.x;
    if (t < count) p[t] = v;
}
__global__ void fill_i32_kernel(int* __restrict__ p, int v, size_t count) {
    size_t t = (size_t)blockIdx.x * blockDim.x + threadIdx.x;
    if (t < count) p[t] = v;
}

// ================= edge passes =================
// pass 1: raw logits + segment max (ordered-int atomicMax, resolves in L2)
__global__ void edge_logit_max_kernel(const long long* __restrict__ es,
                                      const long long* __restrict__ ed,
                                      int E, int Etot,
                                      const float* __restrict__ als,
                                      const float* __restrict__ ald,
                                      float* __restrict__ ew,
                                      int* __restrict__ mmax, int Hh) {
    long long t = (long long)blockIdx.x * blockDim.x + threadIdx.x;
    if (t >= (long long)Etot * Hh) return;
    int e  = (int)(t / Hh);
    int hd = (int)(t - (long long)e * Hh);
    int s = (e < E) ? (int)es[e] : (e - E);   // self-loops appended after real edges
    int d = (e < E) ? (int)ed[e] : (e - E);
    float v = als[(size_t)s * Hh + hd] + ald[(size_t)d * Hh + hd];
    v = (v > 0.f) ? v : NEG_SLOPE * v;        // leaky relu
    ew[t] = v;
    atomicMax(&mmax[(size_t)d * Hh + hd], f2ord(v));
}

// pass 2: exp(e - m[dst]) in place + segment sum
__global__ void edge_expsum_kernel(const long long* __restrict__ ed,
                                   int E, int Etot,
                                   float* __restrict__ ew,
                                   const int* __restrict__ mmax,
                                   float* __restrict__ ssum, int Hh) {
    long long t = (long long)blockIdx.x * blockDim.x + threadIdx.x;
    if (t >= (long long)Etot * Hh) return;
    int e  = (int)(t / Hh);
    int hd = (int)(t - (long long)e * Hh);
    int d = (e < E) ? (int)ed[e] : (e - E);
    float m  = ord2f(mmax[(size_t)d * Hh + hd]);
    float ex = __expf(ew[t] - m);
    ew[t] = ex;
    atomicAdd(&ssum[(size_t)d * Hh + hd], ex);
}

// pass 3: out[dst] += h[src] * alpha   (alpha = ew / (ssum[dst]+eps))
__global__ void edge_agg_kernel(const long long* __restrict__ es,
                                const long long* __restrict__ ed,
                                int E, int Etot,
                                const float* __restrict__ h,
                                const float* __restrict__ ew,
                                const float* __restrict__ ssum,
                                float* __restrict__ out, int Hh, int C) {
    const int F = Hh * C;
    long long t = (long long)blockIdx.x * blockDim.x + threadIdx.x;
    if (t >= (long long)Etot * F) return;
    int e = (int)(t / F);
    int f = (int)(t - (long long)e * F);
    int hd = f / C;
    int s = (e < E) ? (int)es[e] : (e - E);
    int d = (e < E) ? (int)ed[e] : (e - E);
    float alpha = ew[(size_t)e * Hh + hd] / (ssum[(size_t)d * Hh + hd] + 1e-16f);
    atomicAdd(&out[(size_t)d * F + f], h[(size_t)s * F + f] * alpha);
}

// ================= bias (+ optional ReLU) =================
__global__ void bias_act_kernel(const float* __restrict__ acc, const float* __restrict__ b,
                                float* __restrict__ dst, int n, int F, int do_relu) {
    size_t t = (size_t)blockIdx.x * blockDim.x + threadIdx.x;
    if (t >= (size_t)n * F) return;
    int f = (int)(t % F);
    float v = acc[t] + b[f];
    if (do_relu) v = fmaxf(v, 0.f);
    dst[t] = v;
}

static inline unsigned nblk(size_t total, int bs) {
    return (unsigned)((total + (size_t)bs - 1) / (size_t)bs);
}

extern "C" void kernel_launch(void* const* d_in, const int* in_sizes, int n_in,
                              void* d_out, int out_size, void* d_ws, size_t ws_size,
                              hipStream_t stream) {
    (void)n_in; (void)out_size; (void)ws_size;

    const float*     x  = (const float*)d_in[0];
    const long long* ei = (const long long*)d_in[1];   // int64 edge_index [2, E]
    const int N    = in_sizes[0] / 256;
    const int E    = in_sizes[1] / 2;
    const int Etot = E + N;                            // + self loops
    const int npad = ((N + 63) / 64) * 64;

    const float* W[3]  = {(const float*)d_in[2], (const float*)d_in[6],  (const float*)d_in[10]};
    const float* AS[3] = {(const float*)d_in[3], (const float*)d_in[7],  (const float*)d_in[11]};
    const float* AD[3] = {(const float*)d_in[4], (const float*)d_in[8],  (const float*)d_in[12]};
    const float* BI[3] = {(const float*)d_in[5], (const float*)d_in[9],  (const float*)d_in[13]};
    const int FOUT[3]  = {256, 256, in_sizes[10] / 256};
    const int HH[3]    = {8, 8, 1};

    // ---- workspace layout ----
    char* ws = (char*)d_ws;
    size_t off = 0;
    auto alloc = [&](size_t bytes) -> void* {
        void* p = ws + off;
        off = (off + bytes + 255) & ~(size_t)255;
        return p;
    };
    __bf16* Xb = (__bf16*)alloc((size_t)npad * 256 * 2);   // bf16 features
    __bf16* Wt = (__bf16*)alloc((size_t)256 * 256 * 2);    // bf16 transposed weight
    float*  Hb = (float*)alloc((size_t)N * 256 * 4);       // post-GEMM features
    float*  Ob = (float*)alloc((size_t)N * 256 * 4);       // aggregated output
    float* als = (float*)alloc((size_t)N * 8 * 4);
    float* ald = (float*)alloc((size_t)N * 8 * 4);
    int*   mmx = (int*)alloc((size_t)N * 8 * 4);
    float* ssm = (float*)alloc((size_t)N * 8 * 4);
    float* ew  = (float*)alloc((size_t)Etot * 8 * 4);      // per-edge logits/exp

    const long long* esrc = ei;
    const long long* edst = ei + E;

    const float* feat = x;
    for (int l = 0; l < 3; ++l) {
        const int fout = FOUT[l], Hh = HH[l], C = fout / Hh;

        // 1) bf16 conversion (weight stored transposed for WMMA B-fragment loads)
        cvt_x_kernel<<<nblk((size_t)npad * 256, 256), 256, 0, stream>>>(feat, Xb, N, npad);
        cvt_wt_kernel<<<nblk((size_t)fout * 256, 256), 256, 0, stream>>>(W[l], Wt, fout);

        // 2) h = feat @ W  (bf16 WMMA, f32 accumulate)
        {
            dim3 grid(npad / 64, fout / 64);
            gemm_wmma_kernel<<<grid, 256, 0, stream>>>(Xb, Wt, Hb, N, fout);
        }

        // 3) per-node attention coefficients
        attn_coef_kernel<<<nblk((size_t)N * Hh, 256), 256, 0, stream>>>(
            Hb, AS[l], AD[l], als, ald, N, Hh, C);

        // 4) init accumulators
        fill_f32_kernel<<<nblk((size_t)N * fout, 256), 256, 0, stream>>>(Ob, 0.f, (size_t)N * fout);
        fill_i32_kernel<<<nblk((size_t)N * Hh, 256), 256, 0, stream>>>(mmx, (int)0x80000000, (size_t)N * Hh);
        fill_f32_kernel<<<nblk((size_t)N * Hh, 256), 256, 0, stream>>>(ssm, 0.f, (size_t)N * Hh);

        // 5) edge softmax + aggregation
        {
            size_t tot = (size_t)Etot * Hh;
            edge_logit_max_kernel<<<nblk(tot, 256), 256, 0, stream>>>(
                esrc, edst, E, Etot, als, ald, ew, mmx, Hh);
            edge_expsum_kernel<<<nblk(tot, 256), 256, 0, stream>>>(
                edst, E, Etot, ew, mmx, ssm, Hh);
        }
        {
            size_t tot = (size_t)Etot * fout;
            edge_agg_kernel<<<nblk(tot, 256), 256, 0, stream>>>(
                esrc, edst, E, Etot, Hb, ew, ssm, Ob, Hh, C);
        }

        // 6) bias (+ReLU for layers 1-2); final layer writes d_out
        {
            float* dst = (l == 2) ? (float*)d_out : Ob;
            bias_act_kernel<<<nblk((size_t)N * fout, 256), 256, 0, stream>>>(
                Ob, BI[l], dst, N, fout, (l < 2) ? 1 : 0);
        }
        feat = Ob;
    }
}
